// Transformer_depend_80702435492364
// MI455X (gfx1250) — compile-verified, matrix operand order checked by reading, not run on hardware
//
#include <hip/hip_runtime.h>
#include <cstdint>
#include <cstddef>

// ---------------------------------------------------------------------------
// MI455X (gfx1250) transformer forward.
// All GEMM-like work runs on v_wmma_f32_16x16x32_bf16 (fp32->bf16 on the fly,
// f32 accumulate). Compute-bound workload (~0.6 TFLOP vs ~0.25 GB traffic =>
// HBM time ~11us at 23.3 TB/s; WMMA time dominates), so matrix cores are the
// only thing that matters; bf16 is the highest-K dense WMMA for f32 data.
// ---------------------------------------------------------------------------

typedef __attribute__((ext_vector_type(8)))  float   v8f;
typedef __attribute__((ext_vector_type(8)))  __bf16  v8bf;
typedef __attribute__((ext_vector_type(16))) __bf16  v16bf;

// Two fp32 -> packed bf16x2 (round-half-up) in 3 VALU ops via v_perm_b32.
static __device__ __forceinline__ unsigned int f2bf2(float a, float b) {
  unsigned int ua = __float_as_uint(a) + 0x8000u;
  unsigned int ub = __float_as_uint(b) + 0x8000u;
  // out bytes: [0]=ua[2] [1]=ua[3] [2]=ub[2] [3]=ub[3]
  return __builtin_amdgcn_perm(ub, ua, 0x07060302u);
}

static __device__ __forceinline__ float gelu_exact(float x) {
  return 0.5f * x * (1.0f + erff(x * 0.7071067811865476f));
}

// Gather a 16x32 bf16 fragment (A layout / B-transposed layout) from LDS.
// rowp points at this lane's 32-element row; per CDNA5 ISA:
//   elems 0..7  -> K = half*8 + e        (bytes: rowp + half*8,    16B chunk)
//   elems 8..15 -> K = 16 + half*8 + e-8 (bytes: rowp + 16+half*8, 16B chunk)
static __device__ __forceinline__ v16bf ldfrag(const unsigned short* rowp, int half) {
  const v8bf* p0 = reinterpret_cast<const v8bf*>(rowp + (half << 3));
  const v8bf* p1 = reinterpret_cast<const v8bf*>(rowp + 16 + (half << 3));
  v8bf lo = *p0, hi = *p1;
  return __builtin_shufflevector(lo, hi, 0,1,2,3,4,5,6,7,8,9,10,11,12,13,14,15);
}

static __device__ __forceinline__ v8f wmma_bf16(v16bf a, v16bf b, v8f c) {
  return __builtin_amdgcn_wmma_f32_16x16x32_bf16(false, a, false, b, (short)0, c,
                                                 false, false);
}

// ds_swizzle xor-reductions (wave32, EXEC all ones at these points).
#define SWZ(v, imm) __int_as_float(__builtin_amdgcn_ds_swizzle(__float_as_int(v), (imm)))
static __device__ __forceinline__ float rmax16(float v) {   // within 16-lane halves
  v = fmaxf(v, SWZ(v, 0x041F));
  v = fmaxf(v, SWZ(v, 0x081F));
  v = fmaxf(v, SWZ(v, 0x101F));
  v = fmaxf(v, SWZ(v, 0x201F));
  return v;
}
static __device__ __forceinline__ float rsum16(float v) {
  v += SWZ(v, 0x041F); v += SWZ(v, 0x081F);
  v += SWZ(v, 0x101F); v += SWZ(v, 0x201F);
  return v;
}
static __device__ __forceinline__ float rsum32(float v) {
  v += SWZ(v, 0x041F); v += SWZ(v, 0x081F);
  v += SWZ(v, 0x101F); v += SWZ(v, 0x201F); v += SWZ(v, 0x401F);
  return v;
}

// ---------------------------------------------------------------------------
// GEMM: C[M,N] = act(A[M,K] @ B[K,N] + bias).  128x128 block, 8 waves,
// wave = 32x64 sub-tile (2x4 accumulators), K step 32, bf16 WMMA.
// Double-buffered LDS; two-phase (clause-able) global staging via registers.
// M % 128 == 0, N % 128 == 0, K % 32 == 0 (holds for every call site).
// ---------------------------------------------------------------------------
__global__ __launch_bounds__(256, 2)
void k_gemm(const float* __restrict__ A, int lda,
            const float* __restrict__ B, int ldb,
            const float* __restrict__ bias,
            float* __restrict__ C, int ldc,
            int M, int N, int K, int act) {
  __shared__ __align__(16) unsigned short As[2][128 * 32];  // row-major [m][k]
  __shared__ __align__(16) unsigned short Bs[2][128 * 32];  // transposed [n][k]

  const int tid  = threadIdx.x;
  const int lane = tid & 31;
  const int wave = tid >> 5;
  const int half = lane >> 4;
  const int l15  = lane & 15;
  const int m0   = blockIdx.y * 128;
  const int n0   = blockIdx.x * 128;
  const int wm   = wave & 3;   // 4 row groups of 32
  const int wn   = wave >> 2;  // 2 col groups of 64

  float4 ar[4], br[4];
  // Phase 1: issue all global loads for a K tile (clauses, single wait).
  auto loadTiles = [&](int k0) {
#pragma unroll
    for (int j = 0; j < 4; ++j) {
      const int q = tid + j * 256;                 // 1024 quads of A (128x32)
      const int r = q >> 3, c = (q & 7) << 2;
      ar[j] = *reinterpret_cast<const float4*>(A + (size_t)(m0 + r) * lda + k0 + c);
    }
#pragma unroll
    for (int j = 0; j < 4; ++j) {
      const int q = tid + j * 256;                 // 1024 quads of B (32x128)
      const int kk = q >> 5, n = (q & 31) << 2;
      br[j] = *reinterpret_cast<const float4*>(B + (size_t)(k0 + kk) * ldb + n0 + n);
    }
  };
  // Phase 2: convert (v_perm pack) + LDS stores; co-executes with WMMAs.
  auto storeTiles = [&](int buf) {
#pragma unroll
    for (int j = 0; j < 4; ++j) {
      const int q = tid + j * 256;
      const int r = q >> 3, c = (q & 7) << 2;
      uint2 w;
      w.x = f2bf2(ar[j].x, ar[j].y);
      w.y = f2bf2(ar[j].z, ar[j].w);
      *reinterpret_cast<uint2*>(&As[buf][r * 32 + c]) = w;
    }
#pragma unroll
    for (int j = 0; j < 4; ++j) {
      const int q = tid + j * 256;
      const int kk = q >> 5, n = (q & 31) << 2;
      const unsigned int w0 = f2bf2(br[j].x, br[j].y);
      const unsigned int w1 = f2bf2(br[j].z, br[j].w);
      Bs[buf][(n + 0) * 32 + kk] = (unsigned short)w0;
      Bs[buf][(n + 1) * 32 + kk] = (unsigned short)(w0 >> 16);
      Bs[buf][(n + 2) * 32 + kk] = (unsigned short)w1;
      Bs[buf][(n + 3) * 32 + kk] = (unsigned short)(w1 >> 16);
    }
  };

  const v8f vzero = {0.f, 0.f, 0.f, 0.f, 0.f, 0.f, 0.f, 0.f};
  v8f acc[2][4];
#pragma unroll
  for (int i = 0; i < 2; ++i)
#pragma unroll
    for (int j = 0; j < 4; ++j) acc[i][j] = vzero;

  loadTiles(0);
  storeTiles(0);
  __syncthreads();

  int cur = 0;
  for (int k0 = 0; k0 < K; k0 += 32) {
    const bool more = (k0 + 32 < K);
    if (more) loadTiles(k0 + 32);   // global loads in flight during compute

    v16bf af[2], bfr[4];
#pragma unroll
    for (int rt = 0; rt < 2; ++rt)
      af[rt] = ldfrag(&As[cur][(wm * 32 + rt * 16 + l15) * 32], half);
#pragma unroll
    for (int ct = 0; ct < 4; ++ct)
      bfr[ct] = ldfrag(&Bs[cur][(wn * 64 + ct * 16 + l15) * 32], half);
#pragma unroll
    for (int rt = 0; rt < 2; ++rt)
#pragma unroll
      for (int ct = 0; ct < 4; ++ct)
        acc[rt][ct] = wmma_bf16(af[rt], bfr[ct], acc[rt][ct]);

    if (more) storeTiles(cur ^ 1);  // pack VALU overlaps in-flight WMMAs
    __syncthreads();
    cur ^= 1;
  }

  // Epilogue: C layout per ISA -> row = v + half*8, col = l15 within tile.
#pragma unroll
  for (int rt = 0; rt < 2; ++rt) {
#pragma unroll
    for (int ct = 0; ct < 4; ++ct) {
      const int colg = n0 + wn * 64 + ct * 16 + l15;
      const float bv = bias ? bias[colg] : 0.0f;
#pragma unroll
      for (int v = 0; v < 8; ++v) {
        const int rowg = m0 + wm * 32 + rt * 16 + v + half * 8;
        float x = acc[rt][ct][v] + bv;
        if (act) x = gelu_exact(x);
        C[(size_t)rowg * ldc + colg] = x;
      }
    }
  }
}

// ---------------------------------------------------------------------------
// Flash attention, Dh=64, 8 heads. Block = 128 threads (4 waves), each wave
// owns 16 query rows; 64 q rows per block; 32 keys per step. grid=(Tq/64,8,B).
// Q/K/V/O are [rows, ld] fp32 with per-head column offset col0 + h*64.
// ---------------------------------------------------------------------------
__global__ __launch_bounds__(128, 2)
void k_attn(const float* __restrict__ Q, int ldq, int qcol0, int Tq,
            const float* __restrict__ Kp, const float* __restrict__ Vp,
            int ldkv, int kcol0, int vcol0, int Tk,
            float* __restrict__ O, int ldo, int ocol0) {
  __shared__ __align__(16) unsigned short Ks[32 * 64];       // [key][d]
  __shared__ __align__(16) unsigned short Vt[64 * 32];       // [d][key]
  __shared__ __align__(16) unsigned short Pb[4 * 16 * 32];   // per-wave P tile

  const int tid  = threadIdx.x;
  const int lane = tid & 31;
  const int wave = tid >> 5;
  const int half = lane >> 4;
  const int l15  = lane & 15;
  const int h    = blockIdx.y;
  const int z    = blockIdx.z;
  const int qbase = blockIdx.x * 64 + wave * 16;

  // Q fragments (A layout), scaled by Dh^-0.5 = 0.125 before bf16 conversion.
  v16bf qf[2];
  {
    const size_t qrow = (size_t)z * Tq + qbase + l15;
    const float* qp = Q + qrow * ldq + qcol0 + h * 64;
#pragma unroll
    for (int ks = 0; ks < 2; ++ks) {
      union { unsigned int u[8]; v16bf v; } t;
#pragma unroll
      for (int c = 0; c < 2; ++c) {
        const int d0 = ks * 32 + c * 16 + half * 8;
#pragma unroll
        for (int j = 0; j < 4; ++j)
          t.u[c * 4 + j] = f2bf2(qp[d0 + 2 * j] * 0.125f, qp[d0 + 2 * j + 1] * 0.125f);
      }
      qf[ks] = t.v;
    }
  }

  float m[8], l[8];
  v8f o[4];
  const v8f vzero = {0.f, 0.f, 0.f, 0.f, 0.f, 0.f, 0.f, 0.f};
#pragma unroll
  for (int r = 0; r < 8; ++r) { m[r] = -3.0e38f; l[r] = 0.0f; }
#pragma unroll
  for (int d = 0; d < 4; ++d) o[d] = vzero;

  const size_t kvb = (size_t)z * Tk;
  for (int k0 = 0; k0 < Tk; k0 += 32) {
    __syncthreads();
    // Two-phase staging: clause all 8 global_load_b128 first, then convert +
    // LDS stores (K as b64 pairs, V^T scattered b16).
    float4 kr[4], vr[4];
#pragma unroll
    for (int j = 0; j < 4; ++j) {
      const int q = tid + j * 128;                 // 512 quads (32 keys x 64 d)
      const int key = q >> 4, d = (q & 15) << 2;
      const size_t row = (kvb + k0 + key) * (size_t)ldkv;
      kr[j] = *reinterpret_cast<const float4*>(Kp + row + kcol0 + h * 64 + d);
      vr[j] = *reinterpret_cast<const float4*>(Vp + row + vcol0 + h * 64 + d);
    }
#pragma unroll
    for (int j = 0; j < 4; ++j) {
      const int q = tid + j * 128;
      const int key = q >> 4, d = (q & 15) << 2;
      uint2 wk;
      wk.x = f2bf2(kr[j].x, kr[j].y);
      wk.y = f2bf2(kr[j].z, kr[j].w);
      *reinterpret_cast<uint2*>(&Ks[key * 64 + d]) = wk;
      const unsigned int w0 = f2bf2(vr[j].x, vr[j].y);
      const unsigned int w1 = f2bf2(vr[j].z, vr[j].w);
      Vt[(d + 0) * 32 + key] = (unsigned short)w0;
      Vt[(d + 1) * 32 + key] = (unsigned short)(w0 >> 16);
      Vt[(d + 2) * 32 + key] = (unsigned short)w1;
      Vt[(d + 3) * 32 + key] = (unsigned short)(w1 >> 16);
    }
    __syncthreads();

    // S = Q @ K^T for 32 keys (2 col tiles x 2 K-steps).
    v8f s0 = vzero, s1 = vzero;
    s0 = wmma_bf16(qf[0], ldfrag(&Ks[(0 * 16 + l15) * 64 + 0],  half), s0);
    s0 = wmma_bf16(qf[1], ldfrag(&Ks[(0 * 16 + l15) * 64 + 32], half), s0);
    s1 = wmma_bf16(qf[0], ldfrag(&Ks[(1 * 16 + l15) * 64 + 0],  half), s1);
    s1 = wmma_bf16(qf[1], ldfrag(&Ks[(1 * 16 + l15) * 64 + 32], half), s1);

    // Online softmax per row (row r lives in lanes sharing (v=r, half)).
    unsigned short* pw = &Pb[wave * 512];
#pragma unroll
    for (int r = 0; r < 8; ++r) {
      float sm = rmax16(fmaxf(s0[r], s1[r]));
      float mn = fmaxf(m[r], sm);
      float p0 = __expf(s0[r] - mn);
      float p1 = __expf(s1[r] - mn);
      float rs = rsum16(p0 + p1);
      float sc = __expf(m[r] - mn);
      m[r] = mn;
      l[r] = l[r] * sc + rs;
      o[0][r] *= sc; o[1][r] *= sc; o[2][r] *= sc; o[3][r] *= sc;
      const int prow = r + half * 8;
      const unsigned int wp = f2bf2(p0, p1);
      pw[prow * 32 + l15]      = (unsigned short)wp;
      pw[prow * 32 + 16 + l15] = (unsigned short)(wp >> 16);
    }
    // Wave-local LDS RAW: DS ops are in-order per wave; fence compiler + DScnt
    // before the cross-lane fragment re-gather.
    asm volatile("s_wait_dscnt 0" ::: "memory");

    v16bf pf = ldfrag(&pw[l15 * 32], half);
#pragma unroll
    for (int dt = 0; dt < 4; ++dt)
      o[dt] = wmma_bf16(pf, ldfrag(&Vt[(dt * 16 + l15) * 32], half), o[dt]);
  }

  // Write out: C-layout rows; divide by l via v_rcp.
#pragma unroll
  for (int dt = 0; dt < 4; ++dt) {
    const int dcol = ocol0 + h * 64 + dt * 16 + l15;
#pragma unroll
    for (int r = 0; r < 8; ++r) {
      const size_t rowg = (size_t)z * Tq + qbase + r + half * 8;
      O[rowg * ldo + dcol] = o[dt][r] * __builtin_amdgcn_rcpf(l[r]);
    }
  }
}

// ---------------------------------------------------------------------------
// Fused residual + LayerNorm over DIM=512: X = LN(A + X) * g + b.
// One token per wave (16 floats/lane), full-wave ds_swizzle reductions.
// tokens % 8 == 0 at every call site.
// ---------------------------------------------------------------------------
__global__ __launch_bounds__(256, 4)
void k_ln(const float* __restrict__ A, float* __restrict__ X,
          const float* __restrict__ g, const float* __restrict__ b,
          int tokens) {
  const int tid = threadIdx.x, wave = tid >> 5, lane = tid & 31;
  const int t = blockIdx.x * 8 + wave;
  if (t >= tokens) return;
  const size_t base = (size_t)t * 512 + lane * 16;
  const float4* ap = reinterpret_cast<const float4*>(A + base);
  float4*       xp = reinterpret_cast<float4*>(X + base);
  const float4* gp = reinterpret_cast<const float4*>(g + lane * 16);
  const float4* bp = reinterpret_cast<const float4*>(b + lane * 16);

  float4 v[4];
  float s = 0.f, s2 = 0.f;
#pragma unroll
  for (int j = 0; j < 4; ++j) {
    float4 a = ap[j], x = xp[j];
    v[j].x = a.x + x.x; v[j].y = a.y + x.y; v[j].z = a.z + x.z; v[j].w = a.w + x.w;
    s  += v[j].x + v[j].y + v[j].z + v[j].w;
    s2 += v[j].x * v[j].x + v[j].y * v[j].y + v[j].z * v[j].z + v[j].w * v[j].w;
  }
  s = rsum32(s); s2 = rsum32(s2);
  const float mean = s * (1.0f / 512.0f);
  const float var  = s2 * (1.0f / 512.0f) - mean * mean;
  const float rstd = rsqrtf(var + 1e-5f);
#pragma unroll
  for (int j = 0; j < 4; ++j) {
    float4 gg = gp[j], bb = bp[j], y;
    y.x = (v[j].x - mean) * rstd * gg.x + bb.x;
    y.y = (v[j].y - mean) * rstd * gg.y + bb.y;
    y.z = (v[j].z - mean) * rstd * gg.z + bb.z;
    y.w = (v[j].w - mean) * rstd * gg.w + bb.w;
    xp[j] = y;
  }
}

// Layout shuffles between [b,3,n,d] (encoder/out) and per-axis [b,n,d].
__global__ void k_gather(const float* __restrict__ src, float* __restrict__ dst) {
  const long long idx = (long long)blockIdx.x * 256 + threadIdx.x;  // < 3*16*256*512
  const long long a = idx / 2097152;
  const long long r = idx % 2097152;
  const long long bb = r / 131072;
  const long long w = r % 131072;
  dst[idx] = src[(bb * 3 + a) * 131072 + w];
}
__global__ void k_stack(const float* __restrict__ src, float* __restrict__ dst) {
  const long long idx = (long long)blockIdx.x * 256 + threadIdx.x;
  const long long a = idx / 2097152;
  const long long r = idx % 2097152;
  const long long bb = r / 131072;
  const long long w = r % 131072;
  dst[(bb * 3 + a) * 131072 + w] = src[idx];
}

// ---------------------------------------------------------------------------
// Host orchestration.
// Input order (insertion order of setup_inputs): 0=x, 1=emb (unused), then
// enc l=0..3: 13 tensors each (wq wk wv wo bo ln1g ln1b fw1 fb1 fw2 fb2 ln2g ln2b),
// then dec l=0..3: 3 axes x 11 (wqkv wo bo lng lnb fw1 fb1 fw2 fb2 lnfg lnfb)
// + cross 7 (wq wk wv wo bo lng lnb).  n_in = 2 + 52 + 160 = 214.
// ---------------------------------------------------------------------------
extern "C" void kernel_launch(void* const* d_in, const int* in_sizes, int n_in,
                              void* d_out, int out_size, void* d_ws, size_t ws_size,
                              hipStream_t stream) {
  (void)in_sizes; (void)n_in; (void)out_size; (void)ws_size;
  auto in = [&](int i) -> const float* { return (const float*)d_in[i]; };

  // Workspace layout (floats): ~201 MB total.
  float* W    = (float*)d_ws;
  float* ENCX = W;                         // 12288 x 512
  float* H    = ENCX + 6291456;            // 12288 x 2048 (qkv / kv / mlp hidden)
  float* T1   = H + 25165824;              // 12288 x 512
  float* T2   = T1 + 6291456;              // 12288 x 512
  float* Q3   = T2 + 6291456;              // 3 x (4096 x 512)
  const long long QS = 2097152;

  auto gemm = [&](const float* A, int lda, const float* B, int ldb,
                  const float* bias, float* C, int ldc,
                  int M, int N, int K, int act) {
    k_gemm<<<dim3((unsigned)(N / 128), (unsigned)(M / 128)), 256, 0, stream>>>(
        A, lda, B, ldb, bias, C, ldc, M, N, K, act);
  };
  auto attn = [&](const float* Q, int ldq, int qc, int Tq,
                  const float* Kp, const float* Vp, int ldkv, int kc, int vc, int Tk,
                  float* O, int ldo, int oc) {
    k_attn<<<dim3((unsigned)(Tq / 64), 8, 16), 128, 0, stream>>>(
        Q, ldq, qc, Tq, Kp, Vp, ldkv, kc, vc, Tk, O, ldo, oc);
  };
  auto lnres = [&](const float* A, float* X, const float* g, const float* b, int tokens) {
    k_ln<<<(unsigned)(tokens / 8), 256, 0, stream>>>(A, X, g, b, tokens);
  };

  // x -> encoder state ([B,3,N,D] flat == [12288, 512]).
  hipMemcpyAsync(ENCX, d_in[0], (size_t)6291456 * sizeof(float),
                 hipMemcpyDeviceToDevice, stream);

  // ---------------- Encoder ----------------
  for (int lyr = 0; lyr < 4; ++lyr) {
    const int b0 = 2 + 13 * lyr;
    // packed Q|K|V into H (ld 1536)
    gemm(ENCX, 512, in(b0 + 0), 512, nullptr, H + 0,    1536, 12288, 512, 512, 0);
    gemm(ENCX, 512, in(b0 + 1), 512, nullptr, H + 512,  1536, 12288, 512, 512, 0);
    gemm(ENCX, 512, in(b0 + 2), 512, nullptr, H + 1024, 1536, 12288, 512, 512, 0);
    attn(H, 1536, 0, 768, H, H, 1536, 512, 1024, 768, T1, 512, 0);
    gemm(T1, 512, in(b0 + 3), 512, in(b0 + 4), T2, 512, 12288, 512, 512, 0);
    lnres(T2, ENCX, in(b0 + 5), in(b0 + 6), 12288);
    gemm(ENCX, 512, in(b0 + 7), 2048, in(b0 + 8), H, 2048, 12288, 2048, 512, 1);  // GELU
    gemm(H, 2048, in(b0 + 9), 512, in(b0 + 10), T1, 512, 12288, 512, 2048, 0);
    lnres(T1, ENCX, in(b0 + 11), in(b0 + 12), 12288);
  }

  // Split encoder output into per-axis decoder states.
  k_gather<<<24576, 256, 0, stream>>>(ENCX, Q3);

  // ---------------- Decoder ----------------
  for (int lyr = 0; lyr < 4; ++lyr) {
    const int db = 54 + 40 * lyr;
    // Phase A: per-axis self-attention.
    for (int a = 0; a < 3; ++a) {
      const int ab = db + 11 * a;
      float* qa = Q3 + a * QS;
      gemm(qa, 512, in(ab + 0), 1536, nullptr, H, 1536, 4096, 1536, 512, 0);  // wqkv
      attn(H, 1536, 0, 256, H, H, 1536, 512, 1024, 256, T1, 512, 0);
      gemm(T1, 512, in(ab + 1), 512, in(ab + 2), T2, 512, 4096, 512, 512, 0);
      lnres(T2, qa, in(ab + 3), in(ab + 4), 4096);
    }
    // Phase B: shared cross-attention (K/V from encoder output, computed once).
    const int cb = db + 33;
    gemm(ENCX, 512, in(cb + 1), 512, nullptr, H + 0,   1024, 12288, 512, 512, 0);  // wk
    gemm(ENCX, 512, in(cb + 2), 512, nullptr, H + 512, 1024, 12288, 512, 512, 0);  // wv
    for (int a = 0; a < 3; ++a) {
      float* qa = Q3 + a * QS;
      gemm(qa, 512, in(cb + 0), 512, nullptr, T1, 512, 4096, 512, 512, 0);  // wq
      attn(T1, 512, 0, 256, H, H, 1024, 0, 512, 768, T2, 512, 0);
      gemm(T2, 512, in(cb + 3), 512, in(cb + 4), T1, 512, 4096, 512, 512, 0);
      lnres(T1, qa, in(cb + 5), in(cb + 6), 4096);
    }
    // Phase C: per-axis FFN.
    for (int a = 0; a < 3; ++a) {
      const int ab = db + 11 * a;
      float* qa = Q3 + a * QS;
      gemm(qa, 512, in(ab + 5), 2048, in(ab + 6), H, 2048, 4096, 2048, 512, 1);  // GELU
      gemm(H, 2048, in(ab + 7), 512, in(ab + 8), T1, 512, 4096, 512, 2048, 0);
      lnres(T1, qa, in(ab + 9), in(ab + 10), 4096);
    }
  }

  // Stack per-axis states into [B,3,N,D] output.
  k_stack<<<24576, 256, 0, stream>>>(Q3, (float*)d_out);
}
